// MultiHeadedAttention_28956669509733
// MI455X (gfx1250) — matrix operand order repaired and checked by reading, not compiled, not run on hardware
//
#include <hip/hip_runtime.h>

// ---------------------------------------------------------------------------
// MultiHeadedAttention with temporal bias for gfx1250 (MI455X).
// All matmuls run on v_wmma_f32_16x16x32_bf16 (bf16 operands, f32 accum).
// GEMM tiles staged with GLOBAL_LOAD_ASYNC_TO_LDS_B128, double-buffered so the
// DMA for tile i+1 overlaps the WMMA stream for tile i (s_wait_asynccnt 4).
// Workspace layout (bytes), total 56 MB:
//   [ 0,  8M) Q   bf16 (B,H,S,DK)   (pre-scaled by DK^-0.5)
//   [ 8, 16M) K   bf16 (B,H,S,DK)
//   [16, 24M) V^T bf16 (B,H,DK,S)
//   [24, 32M) temporal bias f32 (B,S,S)
//   [32, 40M) context bf16 (B,S,D)
//   [40, 48M) x   bf16 (B*S, D)
//   [48, 50M) wq^T bf16 (N,K); [50,52M) wk^T; [52,54M) wv^T; [54,56M) wproj^T
// ---------------------------------------------------------------------------

#define DEV __device__ __forceinline__

typedef __attribute__((ext_vector_type(16))) __bf16 v16bf;
typedef __attribute__((ext_vector_type(4)))  __bf16 bf16x4;
typedef __attribute__((ext_vector_type(8)))  float  v8f;
typedef __attribute__((ext_vector_type(4)))  float  f32x4;
typedef __attribute__((ext_vector_type(4)))  unsigned int u32x4;
typedef __attribute__((ext_vector_type(4)))  int v4i;

union BF16x16 { u32x4 q[2]; v16bf v; };

#if defined(__has_builtin)
#if __has_builtin(__builtin_amdgcn_global_load_async_to_lds_b128) && \
    __has_builtin(__builtin_amdgcn_s_wait_asynccnt)
#define USE_ASYNC_LDS 1
#endif
#endif

// pointer-to-int4 in global (AS1) / LDS (AS3) address spaces
typedef __attribute__((address_space(1))) v4i* gv4i_p;
typedef __attribute__((address_space(3))) v4i* lv4i_p;
typedef __attribute__((address_space(1))) void* gvoid_p;
typedef __attribute__((address_space(3))) void* lvoid_p;

DEV __bf16 f2bf(float f) { return (__bf16)f; }   // native v_cvt (RNE)

DEV v8f vzero8() {
  v8f z;
#pragma unroll
  for (int e = 0; e < 8; ++e) z[e] = 0.0f;
  return z;
}

// Stage a 64-row x 32-col bf16 tile (row stride gstride elems) into LDS[64][40].
// 2048 elems = 256 x 16B granules; 128 threads issue 2 granules each.
DEV void stage64x32(const __bf16* __restrict__ gbase, size_t gstride,
                    __bf16 (*lds)[40], int tid) {
#ifdef USE_ASYNC_LDS
#pragma unroll
  for (int it = 0; it < 2; ++it) {
    const int g = tid + it * 128;
    const int r = g >> 2, c8 = (g & 3) * 8;
    __builtin_amdgcn_global_load_async_to_lds_b128(
        (gv4i_p)(gvoid_p)(const void*)(gbase + (size_t)r * gstride + c8),
        (lv4i_p)(lvoid_p)(void*)&lds[r][c8], 0, 0);
  }
#else
  for (int i = tid; i < 2048; i += 128) {
    const int r = i >> 5, c = i & 31;
    lds[r][c] = gbase[(size_t)r * gstride + c];
  }
#endif
}

// Allow `PENDING` async granules (the just-issued next tile) to stay in flight.
template <int PENDING>
DEV void wait_async() {
#ifdef USE_ASYNC_LDS
  __builtin_amdgcn_s_wait_asynccnt(PENDING);
#endif
}

// ---------------------------------------------------------------------------
// Prep: x (f32, 4096x1024) -> bf16, vectorized 4 elems/thread.
// ---------------------------------------------------------------------------
__global__ __launch_bounds__(256) void xcvt_kernel(const float* __restrict__ x,
                                                   __bf16* __restrict__ xbf) {
  const int i4 = blockIdx.x * 256 + threadIdx.x;     // 1,048,576 threads
  const f32x4 v = *(const f32x4*)(x + (size_t)i4 * 4);
  bf16x4 o;
#pragma unroll
  for (int j = 0; j < 4; ++j) o[j] = f2bf(v[j]);
  *(bf16x4*)(xbf + (size_t)i4 * 4) = o;
}

// ---------------------------------------------------------------------------
// Prep: weight (K=1024, N=1024) f32 -> bf16 transposed [N][K], tiled via LDS.
// blockIdx.z selects which of the 4 weight matrices.
// ---------------------------------------------------------------------------
__global__ __launch_bounds__(256) void wprep_kernel(
    const float* __restrict__ wq, const float* __restrict__ wk,
    const float* __restrict__ wv, const float* __restrict__ wp,
    __bf16* __restrict__ wqb, __bf16* __restrict__ wkb,
    __bf16* __restrict__ wvb, __bf16* __restrict__ wpb) {
  __shared__ float tile[64][65];
  const int z = blockIdx.z;
  const float* w = (z == 0) ? wq : (z == 1) ? wk : (z == 2) ? wv : wp;
  __bf16* o = (z == 0) ? wqb : (z == 1) ? wkb : (z == 2) ? wvb : wpb;
  const int n0 = blockIdx.x * 64, k0 = blockIdx.y * 64;
  const int tid = threadIdx.x;
  for (int i = tid; i < 4096; i += 256) {
    const int r = i >> 6, c = i & 63;                  // r: k, c: n  (coalesced)
    tile[r][c] = w[(size_t)(k0 + r) * 1024 + n0 + c];
  }
  __syncthreads();
  for (int i = tid; i < 4096; i += 256) {
    const int r = i >> 6, c = i & 63;                  // r: n, c: k  (coalesced)
    o[(size_t)(n0 + r) * 1024 + k0 + c] = f2bf(tile[c][r]);
  }
}

// ---------------------------------------------------------------------------
// Temporal bias MLP: bias[b,q,k] = W2 . leaky(W1 * (1/log(e+t)) + b1) + b2
// ---------------------------------------------------------------------------
__global__ __launch_bounds__(256) void bias_mlp_kernel(
    const float* __restrict__ tm, const float* __restrict__ w1,
    const float* __restrict__ b1, const float* __restrict__ w2,
    const float* __restrict__ b2, float* __restrict__ biasws, int n)
{
  __shared__ float w1s[64], b1s[64], w2s[64];
  const int tid = threadIdx.x;
  if (tid < 64) { w1s[tid] = w1[tid]; b1s[tid] = b1[tid]; w2s[tid] = w2[tid]; }
  __syncthreads();
  const int idx = blockIdx.x * 256 + tid;
  if (idx >= n) return;
  const float t = 1.0f / logf(2.7182818284590452f + tm[idx]);
  float acc = b2[0];
#pragma unroll 8
  for (int j = 0; j < 64; ++j) {
    float h = fmaf(t, w1s[j], b1s[j]);
    h = (h >= 0.0f) ? h : 0.2f * h;
    acc = fmaf(h, w2s[j], acc);
  }
  biasws[idx] = acc;
}

// ---------------------------------------------------------------------------
// QKV projection GEMM: xbf (4096x1024 bf16) @ w^T-tiles + bias.
// blockIdx.z: 0 = Q (scaled, (B,H,S,DK)), 1 = K ((B,H,S,DK)), 2 = V ((B,H,DK,S))
// block = 128 threads (4 waves); block tile 64(M) x 64(N); K-step 32.
// Double-buffered async-LDS pipeline: DMA(i+1) overlaps WMMA(i).
// ---------------------------------------------------------------------------
__global__ __launch_bounds__(128) void gemm_qkv_kernel(
    const __bf16* __restrict__ xbf,
    const __bf16* __restrict__ wqb, const __bf16* __restrict__ wkb,
    const __bf16* __restrict__ wvb,
    const float* __restrict__ bq, const float* __restrict__ bk,
    const float* __restrict__ bv,
    __bf16* __restrict__ qws, __bf16* __restrict__ kws, __bf16* __restrict__ vtws)
{
  __shared__ __bf16 As[2][64][40];   // [buf][row m][k] (80B rows, 16B chunks)
  __shared__ __bf16 Ws[2][64][40];   // [buf][col n][k]

  const int mode = blockIdx.z;
  const __bf16* wt   = (mode == 0) ? wqb : (mode == 1) ? wkb : wvb;
  const float*  bias = (mode == 0) ? bq : (mode == 1) ? bk : bv;
  const int n0 = blockIdx.x * 64;
  const int m0 = blockIdx.y * 64;
  const int tid = threadIdx.x;
  const int wave = tid >> 5, lane = tid & 31;
  const int laneM = lane & 15, half = lane >> 4;

  const __bf16* aG = xbf + (size_t)m0 * 1024;
  const __bf16* wG = wt + (size_t)n0 * 1024;

  v8f acc[4];
#pragma unroll
  for (int t = 0; t < 4; ++t) acc[t] = vzero8();

  stage64x32(aG, 1024, As[0], tid);            // prologue: tile 0
  stage64x32(wG, 1024, Ws[0], tid);

  for (int k0 = 0; k0 < 1024; k0 += 32) {
    const int p = (k0 >> 5) & 1;
    const bool more = (k0 + 32 < 1024);
    if (more) {                                // issue DMA for tile i+1
      stage64x32(aG + k0 + 32, 1024, As[1 - p], tid);
      stage64x32(wG + k0 + 32, 1024, Ws[1 - p], tid);
      wait_async<4>();                         // tile i done; i+1 in flight
      const int r = (tid >> 1), c = (tid & 1) * 16;
      __builtin_prefetch(aG + (size_t)r * 1024 + k0 + 64 + c, 0, 1);
      __builtin_prefetch(wG + (size_t)r * 1024 + k0 + 64 + c, 0, 1);
    } else {
      wait_async<0>();
    }
    __syncthreads();                           // all waves' tile i staged

    BF16x16 a;
    const int arow = wave * 16 + laneM, ab = half * 16;
    a.q[0] = *(const u32x4*)&As[p][arow][ab];
    a.q[1] = *(const u32x4*)&As[p][arow][ab + 8];
#pragma unroll
    for (int t = 0; t < 4; ++t) {
      BF16x16 bm;
      const int nrow = t * 16 + laneM, kb = half * 16;
      bm.q[0] = *(const u32x4*)&Ws[p][nrow][kb];
      bm.q[1] = *(const u32x4*)&Ws[p][nrow][kb + 8];
      acc[t] = __builtin_amdgcn_wmma_f32_16x16x32_bf16(
          false, a.v, false, bm.v, (short)0, acc[t], false, false);
    }
    __syncthreads();                           // reads done before buf reuse
  }

#pragma unroll
  for (int t = 0; t < 4; ++t) {
    const int ng = n0 + t * 16 + laneM;     // global column in D
    const int hh = ng >> 6, dk = ng & 63;
    const float bvv = bias[ng];
#pragma unroll
    for (int e = 0; e < 8; ++e) {
      const int m = m0 + wave * 16 + e + half * 8;   // global row in B*S
      const int bb = m >> 9, ss = m & 511;
      const float val = acc[t][e] + bvv;
      if (mode == 0) {
        qws[(((size_t)bb * 16 + hh) * 512 + ss) * 64 + dk] = f2bf(val * 0.125f);
      } else if (mode == 1) {
        kws[(((size_t)bb * 16 + hh) * 512 + ss) * 64 + dk] = f2bf(val);
      } else {
        vtws[(((size_t)bb * 16 + hh) * 64 + dk) * 512 + ss] = f2bf(val);
      }
    }
  }
}

// ---------------------------------------------------------------------------
// Fused attention: scores (WMMA) + temporal bias + padding/causal masks +
// online softmax + P@V (WMMA). block = 64 threads (2 waves), each wave owns
// 16 query rows; 32 keys per iteration.
// ---------------------------------------------------------------------------
__global__ __launch_bounds__(64) void attn_kernel(
    const __bf16* __restrict__ qws, const __bf16* __restrict__ kws,
    const __bf16* __restrict__ vtws, const float* __restrict__ biasws,
    const int* __restrict__ pmask, __bf16* __restrict__ ctxws)
{
  __shared__ float pbuf[2][16][33];   // per-wave P tile (16 q x 32 keys)

  const int tid = threadIdx.x;
  const int wave = tid >> 5, lane = tid & 31;
  const int laneM = lane & 15, half = lane >> 4;
  const int bh = blockIdx.y;                  // b*16 + h
  const int b = bh >> 4, h = bh & 15;
  const int qt0 = blockIdx.x * 32 + wave * 16;

  const __bf16* qbase = qws + (size_t)bh * 512 * 64;
  const __bf16* kbase = kws + (size_t)bh * 512 * 64;
  const __bf16* vbase = vtws + (size_t)bh * 64 * 512;

  // Q as A-fragments for the whole kernel (DK=64 -> two K=32 chunks)
  v16bf aq[2];
  {
    const __bf16* qrow = qbase + (size_t)(qt0 + laneM) * 64;
#pragma unroll
    for (int c = 0; c < 2; ++c) {
      BF16x16 u;
      u.q[0] = *(const u32x4*)(qrow + c * 32 + half * 16);
      u.q[1] = *(const u32x4*)(qrow + c * 32 + half * 16 + 8);
      aq[c] = u.v;
    }
  }

  float mrow[8], lrow[8];
  v8f o[4];
#pragma unroll
  for (int e = 0; e < 8; ++e) { mrow[e] = -1e30f; lrow[e] = 0.0f; }
#pragma unroll
  for (int t = 0; t < 4; ++t) o[t] = vzero8();

  const int nk = (qt0 + 15) / 32 + 1;         // causal: keys <= qt0+15
  for (int kt = 0; kt < nk; ++kt) {
    const int key0 = kt * 32;

    // ---- scores: S = Q @ K^T (pre-scaled Q), 2 tiles of 16 keys ----
    v8f sc[2];
#pragma unroll
    for (int t = 0; t < 2; ++t) {
      const int key = key0 + t * 16 + laneM;
      const __bf16* krow = kbase + (size_t)key * 64;
      v8f z = vzero8();
#pragma unroll
      for (int c = 0; c < 2; ++c) {
        BF16x16 u;
        u.q[0] = *(const u32x4*)(krow + c * 32 + half * 16);
        u.q[1] = *(const u32x4*)(krow + c * 32 + half * 16 + 8);
        z = __builtin_amdgcn_wmma_f32_16x16x32_bf16(
            false, aq[c], false, u.v, (short)0, z, false, false);
      }
      // bias + padding + causal masks (D layout: row = qt0+e+half*8, col = key)
      const int pm = pmask[b * 512 + key];
#pragma unroll
      for (int e = 0; e < 8; ++e) {
        const int row = qt0 + e + half * 8;
        float v = z[e] + biasws[((size_t)b * 512 + row) * 512 + key];
        if (key > row || pm == 0) v = -1e30f;
        z[e] = v;
      }
      sc[t] = z;
    }

    // ---- online softmax update (row reductions across the 16-lane half) ----
#pragma unroll
    for (int e = 0; e < 8; ++e) {
      float rm = fmaxf(sc[0][e], sc[1][e]);
      rm = fmaxf(rm, __shfl_xor(rm, 1, 32));
      rm = fmaxf(rm, __shfl_xor(rm, 2, 32));
      rm = fmaxf(rm, __shfl_xor(rm, 4, 32));
      rm = fmaxf(rm, __shfl_xor(rm, 8, 32));
      const float mn = fmaxf(mrow[e], rm);
      const float scale = __expf(mrow[e] - mn);
      const float p0 = __expf(sc[0][e] - mn);
      const float p1 = __expf(sc[1][e] - mn);
      float rs = p0 + p1;
      rs += __shfl_xor(rs, 1, 32);
      rs += __shfl_xor(rs, 2, 32);
      rs += __shfl_xor(rs, 4, 32);
      rs += __shfl_xor(rs, 8, 32);
      lrow[e] = lrow[e] * scale + rs;
      mrow[e] = mn;
#pragma unroll
      for (int t = 0; t < 4; ++t) o[t][e] *= scale;
      const int rl = e + half * 8;
      pbuf[wave][rl][laneM]      = p0;
      pbuf[wave][rl][16 + laneM] = p1;
    }
    __threadfence_block();   // order LDS write -> cross-lane LDS read (same wave)

    // ---- P as A-fragment (16 x 32), transposed through LDS ----
    v16bf pa;
#pragma unroll
    for (int j = 0; j < 8; ++j) {
      pa[j]     = f2bf(pbuf[wave][laneM][half * 16 + j]);
      pa[j + 8] = f2bf(pbuf[wave][laneM][half * 16 + 8 + j]);
    }

    // ---- O += P @ V  (V^T rows give contiguous B-fragments) ----
#pragma unroll
    for (int t = 0; t < 4; ++t) {
      const __bf16* vrow = vbase + (size_t)(t * 16 + laneM) * 512;
      BF16x16 u;
      u.q[0] = *(const u32x4*)(vrow + key0 + half * 16);
      u.q[1] = *(const u32x4*)(vrow + key0 + half * 16 + 8);
      o[t] = __builtin_amdgcn_wmma_f32_16x16x32_bf16(
          false, pa, false, u.v, (short)0, o[t], false, false);
    }
  }

  // ---- normalize and store context as bf16 (B,S,D) ----
#pragma unroll
  for (int t = 0; t < 4; ++t) {
    const int dk = t * 16 + laneM;
#pragma unroll
    for (int e = 0; e < 8; ++e) {
      const int q = qt0 + e + half * 8;
      const float val = o[t][e] / lrow[e];
      ctxws[((size_t)b * 512 + q) * 1024 + h * 64 + dk] = f2bf(val);
    }
  }
}

// ---------------------------------------------------------------------------
// Output projection: context(bf16, 4096x1024) @ w_proj^T-tiles + b_proj -> f32.
// Same double-buffered async pipeline as gemm_qkv.
// ---------------------------------------------------------------------------
__global__ __launch_bounds__(128) void gemm_out_kernel(
    const __bf16* __restrict__ ctx, const __bf16* __restrict__ wpb,
    const float* __restrict__ bias, float* __restrict__ out)
{
  __shared__ __bf16 As[2][64][40];
  __shared__ __bf16 Ws[2][64][40];

  const int n0 = blockIdx.x * 64;
  const int m0 = blockIdx.y * 64;
  const int tid = threadIdx.x;
  const int wave = tid >> 5, lane = tid & 31;
  const int laneM = lane & 15, half = lane >> 4;

  const __bf16* aG = ctx + (size_t)m0 * 1024;
  const __bf16* wG = wpb + (size_t)n0 * 1024;

  v8f acc[4];
#pragma unroll
  for (int t = 0; t < 4; ++t) acc[t] = vzero8();

  stage64x32(aG, 1024, As[0], tid);
  stage64x32(wG, 1024, Ws[0], tid);

  for (int k0 = 0; k0 < 1024; k0 += 32) {
    const int p = (k0 >> 5) & 1;
    const bool more = (k0 + 32 < 1024);
    if (more) {
      stage64x32(aG + k0 + 32, 1024, As[1 - p], tid);
      stage64x32(wG + k0 + 32, 1024, Ws[1 - p], tid);
      wait_async<4>();
      const int r = (tid >> 1), c = (tid & 1) * 16;
      __builtin_prefetch(aG + (size_t)r * 1024 + k0 + 64 + c, 0, 1);
      __builtin_prefetch(wG + (size_t)r * 1024 + k0 + 64 + c, 0, 1);
    } else {
      wait_async<0>();
    }
    __syncthreads();

    BF16x16 a;
    const int arow = wave * 16 + laneM, ab = half * 16;
    a.q[0] = *(const u32x4*)&As[p][arow][ab];
    a.q[1] = *(const u32x4*)&As[p][arow][ab + 8];
#pragma unroll
    for (int t = 0; t < 4; ++t) {
      BF16x16 bm;
      const int nrow = t * 16 + laneM, kb = half * 16;
      bm.q[0] = *(const u32x4*)&Ws[p][nrow][kb];
      bm.q[1] = *(const u32x4*)&Ws[p][nrow][kb + 8];
      acc[t] = __builtin_amdgcn_wmma_f32_16x16x32_bf16(
          false, a.v, false, bm.v, (short)0, acc[t], false, false);
    }
    __syncthreads();
  }

#pragma unroll
  for (int t = 0; t < 4; ++t) {
    const int ng = n0 + t * 16 + laneM;
    const float bvv = bias[ng];
#pragma unroll
    for (int e = 0; e < 8; ++e) {
      const int m = m0 + wave * 16 + e + half * 8;
      out[(size_t)m * 1024 + ng] = acc[t][e] + bvv;
    }
  }
}

// ---------------------------------------------------------------------------
extern "C" void kernel_launch(void* const* d_in, const int* in_sizes, int n_in,
                              void* d_out, int out_size, void* d_ws, size_t ws_size,
                              hipStream_t stream) {
  const float* x      = (const float*)d_in[0];
  const int*   pmask  = (const int*)  d_in[1];
  const float* tm     = (const float*)d_in[2];
  const float* wq     = (const float*)d_in[3];
  const float* bq     = (const float*)d_in[4];
  const float* wk     = (const float*)d_in[5];
  const float* bk     = (const float*)d_in[6];
  const float* wv     = (const float*)d_in[7];
  const float* bv     = (const float*)d_in[8];
  const float* w_proj = (const float*)d_in[9];
  const float* b_proj = (const float*)d_in[10];
  const float* w_tb1  = (const float*)d_in[11];
  const float* b_tb1  = (const float*)d_in[12];
  const float* w_tb2  = (const float*)d_in[13];
  const float* b_tb2  = (const float*)d_in[14];
  float* out = (float*)d_out;

  char* ws = (char*)d_ws;
  __bf16* qws    = (__bf16*)(ws);
  __bf16* kws    = (__bf16*)(ws + ((size_t)8  << 20));
  __bf16* vtws   = (__bf16*)(ws + ((size_t)16 << 20));
  float*  biasws = (float*) (ws + ((size_t)24 << 20));
  __bf16* ctxws  = (__bf16*)(ws + ((size_t)32 << 20));
  __bf16* xbf    = (__bf16*)(ws + ((size_t)40 << 20));
  __bf16* wqb    = (__bf16*)(ws + ((size_t)48 << 20));
  __bf16* wkb    = (__bf16*)(ws + ((size_t)50 << 20));
  __bf16* wvb    = (__bf16*)(ws + ((size_t)52 << 20));
  __bf16* wpb    = (__bf16*)(ws + ((size_t)54 << 20));

  // 0) one-time bf16 conversion of x and transposed-bf16 weights
  xcvt_kernel<<<4096, 256, 0, stream>>>(x, xbf);
  wprep_kernel<<<dim3(16, 16, 4), 256, 0, stream>>>(wq, wk, wv, w_proj,
                                                    wqb, wkb, wvb, wpb);
  // 1) temporal bias MLP: (B,S,S) = 2,097,152 elements
  bias_mlp_kernel<<<8192, 256, 0, stream>>>(tm, w_tb1, b_tb1, w_tb2, b_tb2,
                                            biasws, 8 * 512 * 512);
  // 2) Q/K/V projections (z selects which)
  gemm_qkv_kernel<<<dim3(16, 64, 3), 128, 0, stream>>>(
      xbf, wqb, wkb, wvb, bq, bk, bv, qws, kws, vtws);
  // 3) fused attention: 16 q-tiles of 32 x 128 (b,h) pairs
  attn_kernel<<<dim3(16, 128), 64, 0, stream>>>(qws, kws, vtws, biasws,
                                                pmask, ctxws);
  // 4) output projection
  gemm_out_kernel<<<dim3(16, 64), 128, 0, stream>>>(ctxws, wpb, b_proj, out);
}
